// LSTMForecaster_55499567398917
// MI455X (gfx1250) — compile-verified, matrix operand order checked by reading, not run on hardware
//
#include <hip/hip_runtime.h>
#include <hip/hip_bf16.h>
#include <math.h>

// ---------------------------------------------------------------------------
// LSTM forecaster for MI455X (gfx1250): persistent per-layer scan kernels.
// Per-step GEMM on v_wmma_f32_16x16x32_bf16; fused [Wi;Wh] resident in the
// WGP's 320KB LDS; next-step input pipelined via global_load_async_to_lds
// (layer 1) / early register loads (layer 0) so HBM latency is off the
// sequential critical path.
// ---------------------------------------------------------------------------

typedef __bf16 bf16_t;
typedef __attribute__((ext_vector_type(16))) __bf16 v16bf;
typedef __attribute__((ext_vector_type(8)))  __bf16 v8bf;
typedef __attribute__((ext_vector_type(8)))  float  v8f;

#define LSTM_H   128
#define LSTM_G   512     // 4*H gate width
#define SEQ_T    1024
#define BATCH    512
#define BTILE    16      // batch rows per workgroup (= WMMA M)
#define NTHR     256     // 8 waves of 32

__device__ __forceinline__ float sigf(float x) { return 1.0f / (1.0f + __expf(-x)); }

union ABfrag { v16bf v; v8bf h[2]; };

// Async copy 16B global -> LDS (CDNA5 GLOBAL_LOAD_ASYNC_TO_LDS_B128, GV mode).
__device__ __forceinline__ void async_ld_b128(const void* gsrc, void* ldst) {
    unsigned lds = (unsigned)(size_t)ldst;                 // generic addr[31:0] == LDS offset
    unsigned long long ga = (unsigned long long)(size_t)gsrc;
    asm volatile("global_load_async_to_lds_b128 %0, %1, off"
                 :: "v"(lds), "v"(ga) : "memory");
}
__device__ __forceinline__ void wait_async0() {
    asm volatile("s_wait_asynccnt 0x0" ::: "memory");
}

// One workgroup owns BTILE batch rows and scans all SEQ_T steps.
//   Xl[2] : BTILE x DXp  bf16  (double-buffered x_t slab, staged 1 step early)
//   Hl    : BTILE x HP   bf16  (h_{t-1}, rewritten in place each step)
//   Wl    : G x KAp      bf16  (fused [Wi;Wh], column-major over K)
//   gates = [x_t , h] @ W (+bias) : BTILE x G, f32 accum via WMMA
template<int DX, bool IN_BF16, bool WRITE_SEQ>
__global__ __launch_bounds__(NTHR, 1)
void lstm_scan_kernel(const void* __restrict__ xin,
                      const float* __restrict__ Wi,
                      const float* __restrict__ Wh,
                      const float* __restrict__ bias,
                      bf16_t* __restrict__ hseq_out,
                      float*  __restrict__ hfin_out)
{
    constexpr int KA  = DX + LSTM_H;   // fused K (160 or 256)
    constexpr int KAp = KA + 8;        // W column stride (bank-conflict padding)
    constexpr int DXp = DX + 8;        // x slab row stride
    constexpr int HP  = LSTM_H + 8;    // h slab row stride

    extern __shared__ __align__(16) char smem[];
    bf16_t* Wl = (bf16_t*)smem;                          // [G][KAp]
    bf16_t* Xl = Wl + (size_t)LSTM_G * KAp;              // [2][BTILE][DXp]
    bf16_t* Hl = Xl + (size_t)2 * BTILE * DXp;           // [BTILE][HP]
    float*  gl = (float*)(Hl + (size_t)BTILE * HP);      // [BTILE][G]
    float*  cl = gl + (size_t)BTILE * LSTM_G;            // [BTILE][H]
    float*  bl = cl + (size_t)BTILE * LSTM_H;            // [G]

    const int tid = threadIdx.x;
    const int b0  = blockIdx.x * BTILE;

    // ---- one-time: repack fused weights to bf16 column-major, load bias ----
    for (int idx = tid; idx < KA * LSTM_G; idx += NTHR) {
        int g = idx & (LSTM_G - 1);
        int k = idx >> 9;   // G == 512
        float w = (k < DX) ? Wi[(size_t)k * LSTM_G + g]
                           : Wh[(size_t)(k - DX) * LSTM_G + g];
        Wl[(size_t)g * KAp + k] = (bf16_t)w;
    }
    for (int g = tid; g < LSTM_G; g += NTHR) bl[g] = bias[g];
    for (int idx = tid; idx < BTILE * LSTM_H; idx += NTHR) {
        int m = idx >> 7, j = idx & (LSTM_H - 1);
        cl[idx] = 0.0f;                              // c0 = 0
        Hl[(size_t)m * HP + j] = (bf16_t)0.0f;       // h0 = 0
    }

    // ---- prime x_0 into Xl[0] ----
    const int sm = tid >> 4;             // staging row 0..15
    if (IN_BF16) {
        const int sof = (tid & 15) * 8;  // 8 bf16 per thread (DX==128)
        const bf16_t* src0 = (const bf16_t*)xin + ((size_t)(b0 + sm) * SEQ_T + 0) * DX + sof;
        async_ld_b128(src0, Xl + (size_t)sm * DXp + sof);
        wait_async0();
    } else {
        const int sidx = tid * 2;        // 2 fp32 per thread (DX==32)
        const int fm = sidx >> 5, fof = sidx & (DX - 1);
        float2 v = *(const float2*)((const float*)xin + ((size_t)(b0 + fm) * SEQ_T + 0) * DX + fof);
        Xl[(size_t)fm * DXp + fof]     = (bf16_t)v.x;
        Xl[(size_t)fm * DXp + fof + 1] = (bf16_t)v.y;
    }
    __syncthreads();

    const int lane = tid & 31;
    const int wv   = tid >> 5;       // wave id 0..7 -> N-tiles [4w, 4w+3]
    const int l15  = lane & 15;
    const int half = lane >> 4;      // A layout: lanes 16-31 hold K+8

    const int em  = tid >> 4;        // elementwise: row 0..15
    const int ejb = (tid & 15) * 8;  // elementwise: 8 contiguous hidden units

    float2 xreg = {0.0f, 0.0f};      // layer-0 pipelined input registers

    for (int t = 0; t < SEQ_T; ++t) {
        // ---- issue next-step input fetch (fully overlapped with this step) ----
        if (t + 1 < SEQ_T) {
            if (IN_BF16) {
                const int sof = (tid & 15) * 8;
                const bf16_t* src = (const bf16_t*)xin +
                                    ((size_t)(b0 + sm) * SEQ_T + (t + 1)) * DX + sof;
                async_ld_b128(src, Xl + (size_t)((t + 1) & 1) * BTILE * DXp +
                                   (size_t)sm * DXp + sof);
                if (t + 2 < SEQ_T) __builtin_prefetch(src + DX, 0, 0);
            } else {
                const int sidx = tid * 2;
                const int fm = sidx >> 5, fof = sidx & (DX - 1);
                const float* src = (const float*)xin +
                                   ((size_t)(b0 + fm) * SEQ_T + (t + 1)) * DX + fof;
                xreg = *(const float2*)src;
                if (t + 2 < SEQ_T) __builtin_prefetch(src + DX, 0, 0);
            }
        }

        // ---- gates = [x_t, h] @ W  via WMMA bf16 -> f32 ----
        v8f acc[4] = {};
        const bf16_t* xrow = Xl + (size_t)(t & 1) * BTILE * DXp +
                             (size_t)l15 * DXp + half * 8;
        const bf16_t* hrow = Hl + (size_t)l15 * HP + half * 8;
        #pragma unroll
        for (int kt = 0; kt < KA / 32; ++kt) {
            const bf16_t* ap = (kt < DX / 32) ? (xrow + kt * 32)
                                              : (hrow + (kt - DX / 32) * 32);
            ABfrag a;
            a.h[0] = *(const v8bf*)(ap);        // K 0-7  (lanes16-31: 8-15)
            a.h[1] = *(const v8bf*)(ap + 16);   // K 16-23 (lanes16-31: 24-31)
            #pragma unroll
            for (int n = 0; n < 4; ++n) {
                int col = (wv * 4 + n) * 16 + l15;
                const bf16_t* brow = Wl + (size_t)col * KAp + half * 8 + kt * 32;
                ABfrag b;
                b.h[0] = *(const v8bf*)(brow);
                b.h[1] = *(const v8bf*)(brow + 16);
                acc[n] = __builtin_amdgcn_wmma_f32_16x16x32_bf16(
                            false, a.v, false, b.v, (short)0, acc[n], false, false);
            }
        }
        // scatter D (VGPR r: lanes0-15 -> M=r, lanes16-31 -> M=r+8) to gates LDS
        #pragma unroll
        for (int n = 0; n < 4; ++n) {
            int col = (wv * 4 + n) * 16 + l15;
            #pragma unroll
            for (int r = 0; r < 8; ++r)
                gl[(size_t)(r + half * 8) * LSTM_G + col] = acc[n][r];
        }
        __syncthreads();   // barrier #1: gates ready, all A-side reads done

        // ---- gate nonlinearities, state update, next-step staging ----
        {
            const float* gi = gl + (size_t)em * LSTM_G;
            v8bf hv8;
            #pragma unroll
            for (int jj = 0; jj < 8; ++jj) {
                int j = ejb + jj;
                float ig = sigf(gi[j]                + bl[j]);
                float fg = sigf(gi[LSTM_H + j]       + bl[LSTM_H + j]);
                float gg = tanhf(gi[2 * LSTM_H + j]  + bl[2 * LSTM_H + j]);
                float og = sigf(gi[3 * LSTM_H + j]   + bl[3 * LSTM_H + j]);
                float cn = fg * cl[(size_t)em * LSTM_H + j] + ig * gg;
                cl[(size_t)em * LSTM_H + j] = cn;
                float hv = og * tanhf(cn);
                hv8[jj] = (bf16_t)hv;
                if (!WRITE_SEQ && t == SEQ_T - 1)
                    hfin_out[(size_t)(b0 + em) * LSTM_H + j] = hv;
            }
            *(v8bf*)(Hl + (size_t)em * HP + ejb) = hv8;          // feed back h_t
            if (WRITE_SEQ)
                *(v8bf*)(hseq_out + ((size_t)(b0 + em) * SEQ_T + t) * LSTM_H + ejb) = hv8;

            if (!IN_BF16 && t + 1 < SEQ_T) {                     // layer-0: commit x_{t+1}
                const int sidx = tid * 2;
                const int fm = sidx >> 5, fof = sidx & (DX - 1);
                bf16_t* dst = Xl + (size_t)((t + 1) & 1) * BTILE * DXp +
                              (size_t)fm * DXp + fof;
                dst[0] = (bf16_t)xreg.x;
                dst[1] = (bf16_t)xreg.y;
            }
        }
        if (IN_BF16 && t + 1 < SEQ_T) wait_async0();             // x_{t+1} landed in LDS
        __syncthreads();   // barrier #2: A-side (x_{t+1}, h_t) ready for next step
    }
}

// Tiny MLP head: out[b] = relu(h1[b] @ Wd1 + bd1) @ Wd2 + bd2  (8 MFLOP)
__global__ void lstm_head_kernel(const float* __restrict__ h1,
                                 const float* __restrict__ Wd1,
                                 const float* __restrict__ bd1,
                                 const float* __restrict__ Wd2,
                                 const float* __restrict__ bd2,
                                 float* __restrict__ out)
{
    int b = blockIdx.x * blockDim.x + threadIdx.x;
    if (b >= BATCH) return;
    const float* h = h1 + (size_t)b * LSTM_H;
    float acc = bd2[0];
    for (int k = 0; k < LSTM_H / 2; ++k) {
        float s = bd1[k];
        #pragma unroll 8
        for (int j = 0; j < LSTM_H; ++j)
            s += h[j] * Wd1[(size_t)j * (LSTM_H / 2) + k];   // thread-uniform -> scalar loads
        acc += fmaxf(s, 0.0f) * Wd2[k];
    }
    out[b] = acc;
}

extern "C" void kernel_launch(void* const* d_in, const int* in_sizes, int n_in,
                              void* d_out, int out_size, void* d_ws, size_t ws_size,
                              hipStream_t stream)
{
    (void)in_sizes; (void)n_in; (void)out_size; (void)ws_size;

    const float* x   = (const float*)d_in[0];
    const float* Wi0 = (const float*)d_in[1];
    const float* Wh0 = (const float*)d_in[2];
    const float* b0  = (const float*)d_in[3];
    const float* Wi1 = (const float*)d_in[4];
    const float* Wh1 = (const float*)d_in[5];
    const float* b1  = (const float*)d_in[6];
    const float* Wd1 = (const float*)d_in[7];
    const float* bd1 = (const float*)d_in[8];
    const float* Wd2 = (const float*)d_in[9];
    const float* bd2 = (const float*)d_in[10];
    float* out = (float*)d_out;

    // workspace: layer-0 hidden sequence (bf16, 128MB) + final h1 (fp32, 256KB)
    bf16_t* hseq = (bf16_t*)d_ws;
    float*  h1f  = (float*)((char*)d_ws +
                            (size_t)BATCH * SEQ_T * LSTM_H * sizeof(bf16_t));

    constexpr size_t SMEM_FIX = (size_t)BTILE * LSTM_G * 4          // gates
                              + (size_t)BTILE * LSTM_H * 4          // c
                              + (size_t)LSTM_G * 4                  // bias
                              + (size_t)BTILE * (LSTM_H + 8) * 2;   // h slab
    constexpr size_t smem0 = (size_t)LSTM_G * (32  + LSTM_H + 8) * 2
                           + (size_t)2 * BTILE * (32  + 8) * 2 + SMEM_FIX; // ~217KB
    constexpr size_t smem1 = (size_t)LSTM_G * (128 + LSTM_H + 8) * 2
                           + (size_t)2 * BTILE * (128 + 8) * 2 + SMEM_FIX; // ~319KB

    auto* k0 = lstm_scan_kernel<32,  false, true >;
    auto* k1 = lstm_scan_kernel<128, true,  false>;
    (void)hipFuncSetAttribute((const void*)k0, hipFuncAttributeMaxDynamicSharedMemorySize, (int)smem0);
    (void)hipFuncSetAttribute((const void*)k1, hipFuncAttributeMaxDynamicSharedMemorySize, (int)smem1);

    // layer 0: fp32 x in, writes bf16 hidden sequence
    hipLaunchKernelGGL((lstm_scan_kernel<32, false, true>),
                       dim3(BATCH / BTILE), dim3(NTHR), smem0, stream,
                       (const void*)x, Wi0, Wh0, b0, hseq, (float*)nullptr);
    // layer 1: bf16 hidden sequence in (async-to-LDS pipelined), writes final h
    hipLaunchKernelGGL((lstm_scan_kernel<128, true, false>),
                       dim3(BATCH / BTILE), dim3(NTHR), smem1, stream,
                       (const void*)hseq, Wi1, Wh1, b1, (bf16_t*)nullptr, h1f);
    // dense head
    hipLaunchKernelGGL(lstm_head_kernel, dim3(4), dim3(128), 0, stream,
                       h1f, Wd1, bd1, Wd2, bd2, out);
}
